// TGCN_36086315221314
// MI455X (gfx1250) — matrix operand order repaired and checked by reading, not compiled, not run on hardware
//
#include <hip/hip_runtime.h>
#include <hip/hip_bf16.h>

typedef _Float16 v16h __attribute__((ext_vector_type(16)));
typedef float    v8f  __attribute__((ext_vector_type(8)));

#define SEQ_LEN 15
#define BATCH   32
#define N_NODES 5000
#define N_EDGES 80000
#define IN_FEAT 16
#define HIDDEN  512
#define G3      (3 * HIDDEN)          // 1536
#define SB      (SEQ_LEN * BATCH)     // 480
#define KPAD    5024                  // ceil(5000/32)*32
#define EN      (N_EDGES + N_NODES)   // 85000

__device__ __forceinline__ float sig_(float x) { return 1.0f / (1.0f + __expf(-x)); }

// ---------------- utility kernels ----------------
__global__ void k_zero_f32(float* __restrict__ p, long n) {
    long i = (long)blockIdx.x * blockDim.x + threadIdx.x;
    if (i < n) p[i] = 0.0f;
}

__global__ void k_init_h(float* __restrict__ h, _Float16* __restrict__ h16) {
    int i = blockIdx.x * blockDim.x + threadIdx.x;
    if (i < BATCH * HIDDEN) { h[i] = 0.0f; h16[i] = (_Float16)0.0f; }
}

// degree (col occurrences) over real edges; self-loop folded into dinv
__global__ void k_deg(const int* __restrict__ ei, float* __restrict__ deg) {
    int e = blockIdx.x * blockDim.x + threadIdx.x;
    if (e < N_EDGES) atomicAdd(&deg[ei[N_EDGES + e]], 1.0f);
}
__global__ void k_dinv(float* __restrict__ deg) {
    int n = blockIdx.x * blockDim.x + threadIdx.x;
    if (n < N_NODES) deg[n] = rsqrtf(deg[n] + 1.0f);   // +1 = self loop
}

// ---------------- GCN ----------------
// h1[s,b,n,0:2] = x[s,b,n,:] @ W1[s]   (16 -> 2)
__global__ void k_conv1(const float* __restrict__ x, const float* __restrict__ W1,
                        float* __restrict__ h1) {
    long i = (long)blockIdx.x * blockDim.x + threadIdx.x;   // over SB*N
    if (i >= (long)SB * N_NODES) return;
    int s = (int)(i / ((long)BATCH * N_NODES));
    const float4* xp = reinterpret_cast<const float4*>(x + i * IN_FEAT);
    const float* w = W1 + s * IN_FEAT * 2;
    float a0 = 0.0f, a1 = 0.0f;
#pragma unroll
    for (int q = 0; q < 4; ++q) {
        float4 v = xp[q];
        const float* wp = w + q * 8;
        a0 += v.x * wp[0] + v.y * wp[2] + v.z * wp[4] + v.w * wp[6];
        a1 += v.x * wp[1] + v.y * wp[3] + v.z * wp[5] + v.w * wp[7];
    }
    h1[i * 2] = a0; h1[i * 2 + 1] = a1;
}

// scatter-add, 2 channels: agg[sb,col,:] += norm * h[sb,row,:]
__global__ void k_scatter2(const int* __restrict__ ei, const float* __restrict__ dinv,
                           const float* __restrict__ h, float* __restrict__ agg) {
    long t = (long)blockIdx.x * blockDim.x + threadIdx.x;
    if (t >= (long)EN * SB) return;
    int e  = (int)(t % EN);
    int sb = (int)(t / EN);
    int r, c;
    if (e < N_EDGES) { r = ei[e]; c = ei[N_EDGES + e]; } else { r = c = e - N_EDGES; }
    float nm = dinv[r] * dinv[c];
    long rb = ((long)sb * N_NODES + r) * 2;
    long cb = ((long)sb * N_NODES + c) * 2;
    atomicAdd(&agg[cb],     nm * h[rb]);
    atomicAdd(&agg[cb + 1], nm * h[rb + 1]);
}

// t[s,b,n] = relu(agg1 + b1[s]) @ W2[s]   (2 -> 1)
__global__ void k_relu_conv2(const float* __restrict__ agg1, const float* __restrict__ b1,
                             const float* __restrict__ W2, float* __restrict__ t) {
    long i = (long)blockIdx.x * blockDim.x + threadIdx.x;   // over SB*N
    if (i >= (long)SB * N_NODES) return;
    int s = (int)(i / ((long)BATCH * N_NODES));
    float a = fmaxf(agg1[i * 2]     + b1[s * 2],     0.0f);
    float b = fmaxf(agg1[i * 2 + 1] + b1[s * 2 + 1], 0.0f);
    t[i] = a * W2[s * 2] + b * W2[s * 2 + 1];
}

// scatter-add, 1 channel
__global__ void k_scatter1(const int* __restrict__ ei, const float* __restrict__ dinv,
                           const float* __restrict__ t, float* __restrict__ agg) {
    long q = (long)blockIdx.x * blockDim.x + threadIdx.x;
    if (q >= (long)EN * SB) return;
    int e  = (int)(q % EN);
    int sb = (int)(q / EN);
    int r, c;
    if (e < N_EDGES) { r = ei[e]; c = ei[N_EDGES + e]; } else { r = c = e - N_EDGES; }
    float nm = dinv[r] * dinv[c];
    atomicAdd(&agg[(long)sb * N_NODES + c], nm * t[(long)sb * N_NODES + r]);
}

// g16[sb, 0:KPAD] = f16(tanh(agg2 + b2[s])), zero-padded K tail
__global__ void k_g16(const float* __restrict__ agg2, const float* __restrict__ b2,
                      _Float16* __restrict__ g16) {
    long i = (long)blockIdx.x * blockDim.x + threadIdx.x;   // over SB*KPAD
    if (i >= (long)SB * KPAD) return;
    int sb = (int)(i / KPAD);
    int n  = (int)(i % KPAD);
    float v = 0.0f;
    if (n < N_NODES) v = tanhf(agg2[(long)sb * N_NODES + n] + b2[sb / BATCH]);
    g16[i] = (_Float16)v;
}

// fp32 [rows,kin] -> fp16 [rows,kout] (K zero-padded)
__global__ void k_cvt_pad(const float* __restrict__ w, _Float16* __restrict__ o,
                          int rows, int kin, int kout) {
    long i = (long)blockIdx.x * blockDim.x + threadIdx.x;
    if (i >= (long)rows * kout) return;
    int r = (int)(i / kout), k = (int)(i % kout);
    o[i] = (_Float16)(k < kin ? w[(long)r * kin + k] : 0.0f);
}

// -------- fragment loaders (CDNA5 16-bit WMMA VGPR layouts, ISA 7.12.2) --------
union Frag { v16h h; float4 f[2]; };

// A 16x32 f16: lane holds row lane%16; halves 0-7 = K k0+8*(lane/16)..,
// halves 8-15 = K k0+16+8*(lane/16)..
__device__ __forceinline__ v16h load_a_frag(const _Float16* aRow, int k0, int hs) {
    Frag fa;
    fa.f[0] = *reinterpret_cast<const float4*>(aRow + k0 + hs * 8);
    fa.f[1] = *reinterpret_cast<const float4*>(aRow + k0 + 16 + hs * 8);
    return fa.h;
}
// B 32x16 f16: lane holds column lane%16; halves 0-7 = K k0+8*(lane/16).., halves 8-15 = K k0+16+8*(lane/16)..
// B col n == row n of row-major W[N,K]  -> contiguous loads.
__device__ __forceinline__ v16h load_b_frag(const _Float16* wRow, int k0, int hs) {
    Frag fb;
    fb.f[0] = *reinterpret_cast<const float4*>(wRow + k0 + hs * 8);
    fb.f[1] = *reinterpret_cast<const float4*>(wRow + k0 + 16 + hs * 8);
    return fb.h;
}
// D 16x16 f32: VGPR r -> row 8*(lane/16)+r, col lane%16
__device__ __forceinline__ void store_d_tile(float* __restrict__ D, v8f acc,
                                             int m0, int n0, int N, int hs, int lid) {
    int n = n0 + lid;
#pragma unroll
    for (int r = 0; r < 8; ++r)
        D[(size_t)(m0 + hs * 8 + r) * N + n] = acc[r];
}

// ---- 16x16-tile GEMM (high parallelism; used for small-M recurrent steps) ----
// D[M,N] = A[M,K]f16 . W[N,K]f16^T ; K multiple of 32; one wave per tile.
__global__ void k_wmma_gemm_abt(const _Float16* __restrict__ A,
                                const _Float16* __restrict__ W,
                                float* __restrict__ D,
                                int M, int N, int K) {
    int gwave = (int)((blockIdx.x * (long)blockDim.x + threadIdx.x) >> 5);
    int lane  = threadIdx.x & 31;
    int tilesN = N >> 4;
    int mt = gwave / tilesN;
    int nt = gwave % tilesN;
    if (mt * 16 >= M) return;
    int hs  = lane >> 4;
    int lid = lane & 15;
    const _Float16* aRow = A + (size_t)(mt * 16 + lid) * K;
    const _Float16* wRow = W + (size_t)(nt * 16 + lid) * K;
    v8f acc = {};
    for (int k0 = 0; k0 < K; k0 += 32) {
        v16h fa = load_a_frag(aRow, k0, hs);
        v16h fb = load_b_frag(wRow, k0, hs);
        acc = __builtin_amdgcn_wmma_f32_16x16x32_f16(
            false, fa, false, fb, (short)0, acc, false, false);
    }
    store_d_tile(D, acc, mt * 16, nt * 16, N, hs, lid);
}

// ---- 32x32-register-tile GEMM (fragment reuse; used for the big projections) ----
// 2 A-frags x 2 B-frags -> 4 WMMAs per k-step: 2 b128 loads per WMMA instead of 4.
// Requires M % 32 == 0, N % 32 == 0, K % 32 == 0.
__global__ void k_wmma_gemm_abt_32(const _Float16* __restrict__ A,
                                   const _Float16* __restrict__ W,
                                   float* __restrict__ D,
                                   int M, int N, int K) {
    int gwave = (int)((blockIdx.x * (long)blockDim.x + threadIdx.x) >> 5);
    int lane  = threadIdx.x & 31;
    int tilesN = N >> 5;                     // 32-wide tiles
    int mt = gwave / tilesN;
    int nt = gwave % tilesN;
    if (mt * 32 >= M) return;
    int hs  = lane >> 4;
    int lid = lane & 15;
    const _Float16* aRow0 = A + (size_t)(mt * 32 + lid) * K;
    const _Float16* aRow1 = aRow0 + (size_t)16 * K;
    const _Float16* wRow0 = W + (size_t)(nt * 32 + lid) * K;
    const _Float16* wRow1 = wRow0 + (size_t)16 * K;
    v8f acc00 = {}, acc01 = {}, acc10 = {}, acc11 = {};
    for (int k0 = 0; k0 < K; k0 += 32) {
        v16h fa0 = load_a_frag(aRow0, k0, hs);
        v16h fa1 = load_a_frag(aRow1, k0, hs);
        v16h fb0 = load_b_frag(wRow0, k0, hs);
        v16h fb1 = load_b_frag(wRow1, k0, hs);
        acc00 = __builtin_amdgcn_wmma_f32_16x16x32_f16(false, fa0, false, fb0, (short)0, acc00, false, false);
        acc01 = __builtin_amdgcn_wmma_f32_16x16x32_f16(false, fa0, false, fb1, (short)0, acc01, false, false);
        acc10 = __builtin_amdgcn_wmma_f32_16x16x32_f16(false, fa1, false, fb0, (short)0, acc10, false, false);
        acc11 = __builtin_amdgcn_wmma_f32_16x16x32_f16(false, fa1, false, fb1, (short)0, acc11, false, false);
    }
    store_d_tile(D, acc00, mt * 32,      nt * 32,      N, hs, lid);
    store_d_tile(D, acc01, mt * 32,      nt * 32 + 16, N, hs, lid);
    store_d_tile(D, acc10, mt * 32 + 16, nt * 32,      N, hs, lid);
    store_d_tile(D, acc11, mt * 32 + 16, nt * 32 + 16, N, hs, lid);
}

// ---------------- GRU gate update (one timestep) ----------------
__global__ void k_gru_gate(const float* __restrict__ gi, const float* __restrict__ gh,
                           const float* __restrict__ bih, const float* __restrict__ bhh,
                           float* __restrict__ h, _Float16* __restrict__ h16,
                           float* __restrict__ out_f32, _Float16* __restrict__ out_f16) {
    int i = blockIdx.x * blockDim.x + threadIdx.x;   // over BATCH*HIDDEN
    if (i >= BATCH * HIDDEN) return;
    int b = i / HIDDEN, j = i % HIDDEN;
    const float* gib = gi + (long)b * G3;
    const float* ghb = gh + (long)b * G3;
    float ir = gib[j]              + bih[j];
    float iz = gib[HIDDEN + j]     + bih[HIDDEN + j];
    float in = gib[2 * HIDDEN + j] + bih[2 * HIDDEN + j];
    float hr = ghb[j]              + bhh[j];
    float hz = ghb[HIDDEN + j]     + bhh[HIDDEN + j];
    float hn = ghb[2 * HIDDEN + j] + bhh[2 * HIDDEN + j];
    float r = sig_(ir + hr);
    float z = sig_(iz + hz);
    float n = tanhf(in + r * hn);
    float hw = (1.0f - z) * n + z * h[i];
    h[i] = hw;
    h16[i] = (_Float16)hw;
    if (out_f32) out_f32[i] = hw;
    if (out_f16) out_f16[i] = (_Float16)hw;
}

__global__ void k_copy_hn(const float* __restrict__ h0, const float* __restrict__ h1,
                          float* __restrict__ out) {
    int i = blockIdx.x * blockDim.x + threadIdx.x;
    if (i >= BATCH * HIDDEN) return;
    const int base = SEQ_LEN * BATCH * HIDDEN;      // 245760
    out[base + i] = h0[i];
    out[base + BATCH * HIDDEN + i] = h1[i];
}

// ---------------- host launch ----------------
extern "C" void kernel_launch(void* const* d_in, const int* in_sizes, int n_in,
                              void* d_out, int out_size, void* d_ws, size_t ws_size,
                              hipStream_t stream) {
    const float* x     = (const float*)d_in[0];
    const int*   ei    = (const int*)  d_in[1];
    const float* W1    = (const float*)d_in[2];
    const float* b1    = (const float*)d_in[3];
    const float* W2    = (const float*)d_in[4];
    const float* b2    = (const float*)d_in[5];
    const float* w_ih0 = (const float*)d_in[6];
    const float* w_hh0 = (const float*)d_in[7];
    const float* b_ih0 = (const float*)d_in[8];
    const float* b_hh0 = (const float*)d_in[9];
    const float* w_ih1 = (const float*)d_in[10];
    const float* w_hh1 = (const float*)d_in[11];
    const float* b_ih1 = (const float*)d_in[12];
    const float* b_hh1 = (const float*)d_in[13];
    float* out = (float*)d_out;

    char* ws = (char*)d_ws;
    size_t off = 0;
    auto alloc = [&](size_t bytes) -> void* {
        void* p = ws + off;
        off = (off + bytes + 255) & ~(size_t)255;
        return p;
    };
    float*     deg     = (float*)alloc((size_t)N_NODES * 4);                 // -> dinv in place
    float*     h1      = (float*)alloc((size_t)SB * N_NODES * 2 * 4);        // aliased as t later
    float*     agg1    = (float*)alloc((size_t)SB * N_NODES * 2 * 4);        // aliased as agg2 later
    _Float16*  g16     = (_Float16*)alloc((size_t)SB * KPAD * 2);
    _Float16*  wih0_16 = (_Float16*)alloc((size_t)G3 * KPAD * 2);
    _Float16*  whh0_16 = (_Float16*)alloc((size_t)G3 * HIDDEN * 2);
    _Float16*  wih1_16 = (_Float16*)alloc((size_t)G3 * HIDDEN * 2);
    _Float16*  whh1_16 = (_Float16*)alloc((size_t)G3 * HIDDEN * 2);
    float*     gi0     = (float*)alloc((size_t)SB * G3 * 4);
    float*     gi1     = (float*)alloc((size_t)SB * G3 * 4);
    float*     gh      = (float*)alloc((size_t)BATCH * G3 * 4);
    float*     h0      = (float*)alloc((size_t)BATCH * HIDDEN * 4);
    _Float16*  h0_16   = (_Float16*)alloc((size_t)BATCH * HIDDEN * 2);
    float*     h1s     = (float*)alloc((size_t)BATCH * HIDDEN * 4);
    _Float16*  h1_16   = (_Float16*)alloc((size_t)BATCH * HIDDEN * 2);
    _Float16*  out1_16 = (_Float16*)alloc((size_t)SB * HIDDEN * 2);
    float*     t    = h1;    // alias: h1 dead after scatter2
    float*     agg2 = agg1;  // alias: agg1 dead after relu_conv2

    const long nSBN  = (long)SB * N_NODES;                  // 2.4M
    const long nScat = (long)EN * SB;                       // 40.8M

    // --- GCN normalization ---
    k_zero_f32<<<(N_NODES + 255) / 256, 256, 0, stream>>>(deg, N_NODES);
    k_deg <<<(N_EDGES + 255) / 256, 256, 0, stream>>>(ei, deg);
    k_dinv<<<(N_NODES + 255) / 256, 256, 0, stream>>>(deg);

    // --- conv1 -> propagate -> relu -> conv2 -> propagate -> tanh ---
    k_conv1<<<(unsigned)((nSBN + 255) / 256), 256, 0, stream>>>(x, W1, h1);
    k_zero_f32<<<(unsigned)((nSBN * 2 + 255) / 256), 256, 0, stream>>>(agg1, nSBN * 2);
    k_scatter2<<<(unsigned)((nScat + 255) / 256), 256, 0, stream>>>(ei, deg, h1, agg1);
    k_relu_conv2<<<(unsigned)((nSBN + 255) / 256), 256, 0, stream>>>(agg1, b1, W2, t);
    k_zero_f32<<<(unsigned)((nSBN + 255) / 256), 256, 0, stream>>>(agg2, nSBN);
    k_scatter1<<<(unsigned)((nScat + 255) / 256), 256, 0, stream>>>(ei, deg, t, agg2);
    k_g16<<<(unsigned)(((long)SB * KPAD + 255) / 256), 256, 0, stream>>>(agg2, b2, g16);

    // --- f16 weight staging ---
    k_cvt_pad<<<(unsigned)(((long)G3 * KPAD + 255) / 256), 256, 0, stream>>>(w_ih0, wih0_16, G3, N_NODES, KPAD);
    k_cvt_pad<<<(unsigned)(((long)G3 * HIDDEN + 255) / 256), 256, 0, stream>>>(w_hh0, whh0_16, G3, HIDDEN, HIDDEN);
    k_cvt_pad<<<(unsigned)(((long)G3 * HIDDEN + 255) / 256), 256, 0, stream>>>(w_ih1, wih1_16, G3, HIDDEN, HIDDEN);
    k_cvt_pad<<<(unsigned)(((long)G3 * HIDDEN + 255) / 256), 256, 0, stream>>>(w_hh1, whh1_16, G3, HIDDEN, HIDDEN);

    // --- gi0 = g . w_ih0^T : [480 x 5024] . [5024 x 1536], 32x32 register tiles ---
    {
        int tiles = (SB / 32) * (G3 / 32);                  // 15*48 = 720 waves
        k_wmma_gemm_abt_32<<<tiles / 4, 128, 0, stream>>>(g16, wih0_16, gi0, SB, G3, KPAD);
    }

    // --- GRU layer 0 scan ---
    k_init_h<<<(BATCH * HIDDEN + 255) / 256, 256, 0, stream>>>(h0, h0_16);
    for (int s = 0; s < SEQ_LEN; ++s) {
        int tiles = (BATCH / 16) * (G3 / 16);               // 2*96 = 192 waves
        k_wmma_gemm_abt<<<tiles / 4, 128, 0, stream>>>(h0_16, whh0_16, gh, BATCH, G3, HIDDEN);
        k_gru_gate<<<(BATCH * HIDDEN + 255) / 256, 256, 0, stream>>>(
            gi0 + (long)s * BATCH * G3, gh, b_ih0, b_hh0,
            h0, h0_16, (float*)nullptr, out1_16 + (long)s * BATCH * HIDDEN);
    }

    // --- gi1 = out1 . w_ih1^T : [480 x 512] . [512 x 1536], 32x32 register tiles ---
    {
        int tiles = (SB / 32) * (G3 / 32);                  // 720 waves
        k_wmma_gemm_abt_32<<<tiles / 4, 128, 0, stream>>>(out1_16, wih1_16, gi1, SB, G3, HIDDEN);
    }

    // --- GRU layer 1 scan (writes out2 directly) ---
    k_init_h<<<(BATCH * HIDDEN + 255) / 256, 256, 0, stream>>>(h1s, h1_16);
    for (int s = 0; s < SEQ_LEN; ++s) {
        int tiles = (BATCH / 16) * (G3 / 16);
        k_wmma_gemm_abt<<<tiles / 4, 128, 0, stream>>>(h1_16, whh1_16, gh, BATCH, G3, HIDDEN);
        k_gru_gate<<<(BATCH * HIDDEN + 255) / 256, 256, 0, stream>>>(
            gi1 + (long)s * BATCH * G3, gh, b_ih1, b_hh1,
            h1s, h1_16, out + (long)s * BATCH * HIDDEN, (_Float16*)nullptr);
    }

    // --- hn = stack(h_layer0, h_layer1) ---
    k_copy_hn<<<(BATCH * HIDDEN + 255) / 256, 256, 0, stream>>>(h0, h1s, out);
}
